// GTrans1_74552042324535
// MI455X (gfx1250) — matrix operand order repaired and checked by reading, not compiled, not run on hardware
//
#include <hip/hip_runtime.h>

// ---------------------------------------------------------------------------
// GTrans encoder layer for gfx1250 (MI455X): f16 WMMA GEMMs + flash attention
// B=16, L=1024, D=256, H=4, E=64, F=512
// ---------------------------------------------------------------------------

typedef __attribute__((ext_vector_type(16))) _Float16 v16h;
typedef __attribute__((ext_vector_type(8)))  _Float16 v8h;
typedef __attribute__((ext_vector_type(8)))  float    v8f;

#define WMMA_F16(a, b, c) \
  __builtin_amdgcn_wmma_f32_16x16x32_f16(false, (a), false, (b), (short)0, (c), false, false)

constexpr int Bn = 16;
constexpr int Ln = 1024;
constexpr int Dn = 256;
constexpr int Hn = 4;
constexpr int En = 64;
constexpr int Fn = 512;
constexpr int Mn = Bn * Ln; // 16384 rows

__device__ __forceinline__ v8f zero8() {
  v8f z;
#pragma unroll
  for (int j = 0; j < 8; ++j) z[j] = 0.0f;
  return z;
}

// Assemble a wave32 WMMA A-operand (16x32 f16, M=lane%16) from a row-major row:
// element i -> K = 2*off + i
// where off = 8*(lane>=16) relative to the 32-wide K chunk at `rowp`.
__device__ __forceinline__ v16h load_a16(const _Float16* rowp, int off) {
  v8h lo = *(const v8h*)(rowp + 2 * off);
  v8h hi = *(const v8h*)(rowp + 2 * off + 8);
  v16h r;
#pragma unroll
  for (int i = 0; i < 8; ++i) { r[i] = lo[i]; r[i + 8] = hi[i]; }
  return r;
}

__device__ __forceinline__ float gelu_exact(float v) {
  return 0.5f * v * (1.0f + erff(v * 0.70710678118654752f));
}

// ---------------------------------------------------------------------------
// f32 -> f16 convert
// ---------------------------------------------------------------------------
__global__ __launch_bounds__(256) void k_f32_to_f16(const float* __restrict__ in,
                                                    _Float16* __restrict__ out,
                                                    size_t n) {
  size_t i = (size_t)blockIdx.x * blockDim.x + threadIdx.x;
  if (i < n) out[i] = (_Float16)in[i];
}

// Wt[n*K + k] = (f16) W[k*N + n]   (W is [K,N] row-major)
__global__ __launch_bounds__(256) void k_transpose_f16(const float* __restrict__ W,
                                                       _Float16* __restrict__ Wt,
                                                       int K, int N) {
  size_t i = (size_t)blockIdx.x * blockDim.x + threadIdx.x;
  if (i < (size_t)K * N) {
    int n = (int)(i / K);
    int kk = (int)(i % K);
    Wt[i] = (_Float16)W[(size_t)kk * N + n];
  }
}

// ---------------------------------------------------------------------------
// Generic WMMA GEMM: C[m,n] = A[m,:K] * Bt[n,:K]^T + bias[n], with epilogue.
//  A: f16 row-major [M,K]; Bt: f16 row-major [N,K] (pre-transposed weights)
//  block = 256 threads (8 waves), tile = 128(M) x 64(N), K chunked by 32.
//  mode 0: write f16 to [B,H,L,E]            (q / k)
//  mode 1: write f16 to [B,H,E,L]            (v transposed)
//  mode 2: write f32 (v + bias + xres)       (attn out + residual)
//  mode 3: write f16 gelu(v)                 (FFN1)
//  mode 4: write f32 gelu(v)                 (FFN2)
// ---------------------------------------------------------------------------
__global__ __launch_bounds__(256) void k_gemm_f16(const _Float16* __restrict__ A,
                                                  const _Float16* __restrict__ Bt,
                                                  const float* __restrict__ bias,
                                                  int K, int N, int mode,
                                                  const float* __restrict__ xres,
                                                  void* __restrict__ outp) {
  __shared__ _Float16 As[128 * 32];
  __shared__ _Float16 Bs[64 * 32];

  const int tid  = threadIdx.x;
  const int wave = tid >> 5;
  const int lane = tid & 31;
  const int l16  = lane & 15;
  const int lh   = lane >> 4;
  const int mb   = blockIdx.y * 128;
  const int nb   = blockIdx.x * 64;

  v8f acc[4];
#pragma unroll
  for (int i = 0; i < 4; ++i) acc[i] = zero8();

  for (int kc = 0; kc < K; kc += 32) {
    // cooperative staging: A tile 128x32, B tile 64x32
    {
      int r = tid >> 1, c = (tid & 1) * 16;
      *(v16h*)(&As[r * 32 + c]) = *(const v16h*)(A + (size_t)(mb + r) * K + kc + c);
      int r2 = tid >> 2, c2 = (tid & 3) * 8;
      *(v8h*)(&Bs[r2 * 32 + c2]) = *(const v8h*)(Bt + (size_t)(nb + r2) * K + kc + c2);
    }
    __syncthreads();

    v16h a = load_a16(&As[(wave * 16 + l16) * 32], lh * 8);
#pragma unroll
    for (int nt = 0; nt < 4; ++nt) {
      v16h bv = *(const v16h*)(&Bs[(nt * 16 + l16) * 32 + lh * 16]);
      acc[nt] = WMMA_F16(a, bv, acc[nt]);
    }
    __syncthreads();
  }

  // epilogue (C/D layout: VGPR r -> row r + 8*(lane>=16), col = lane%16)
#pragma unroll
  for (int nt = 0; nt < 4; ++nt) {
    int n = nb + nt * 16 + l16;
    float bn = bias[n];
#pragma unroll
    for (int r = 0; r < 8; ++r) {
      int m = mb + wave * 16 + lh * 8 + r;
      float v = acc[nt][r] + bn;
      if (mode == 0 || mode == 1) {
        int bb = m >> 10, l = m & 1023, hh = n >> 6, e = n & 63;
        size_t idx = (mode == 0)
            ? ((((size_t)(bb * Hn + hh)) * Ln + l) * En + e)
            : ((((size_t)(bb * Hn + hh)) * En + e) * Ln + l);
        ((_Float16*)outp)[idx] = (_Float16)v;
      } else if (mode == 2) {
        size_t idx = (size_t)m * Dn + n;
        ((float*)outp)[idx] = v + xres[idx];
      } else if (mode == 3) {
        ((_Float16*)outp)[(size_t)m * N + n] = (_Float16)gelu_exact(v);
      } else {
        ((float*)outp)[(size_t)m * N + n] = gelu_exact(v);
      }
    }
  }
}

// ---------------------------------------------------------------------------
// Flash attention: 1 wave = 16 query rows of one (b,h); block = 4 waves.
// grid = B*H*(L/64) = 1024.  q,k: [B,H,L,E] f16, vT: [B,H,E,L] f16.
// ctx out: [B,L,D] f16 (heads re-interleaved).
// ---------------------------------------------------------------------------
__global__ __launch_bounds__(128) void k_attention(const _Float16* __restrict__ q,
                                                   const _Float16* __restrict__ k,
                                                   const _Float16* __restrict__ vT,
                                                   const float* __restrict__ bias,
                                                   _Float16* __restrict__ ctx) {
  __shared__ _Float16 pshm[4][16 * 32];

  const int tid  = threadIdx.x;
  const int wave = tid >> 5;
  const int lane = tid & 31;
  const int l16  = lane & 15;
  const int lh   = lane >> 4;
  const int aoff = lh * 8;

  const int bh = blockIdx.x >> 4;   // 0 .. B*H-1
  const int rb = blockIdx.x & 15;   // row block within L
  const int b  = bh >> 2;
  const int hh = bh & 3;
  const int q0 = rb * 64 + wave * 16;

  const _Float16* qp = q  + (size_t)bh * Ln * En;
  const _Float16* kp = k  + (size_t)bh * Ln * En;
  const _Float16* vp = vT + (size_t)bh * En * Ln;
  const float* bp_base = bias + (size_t)bh * Ln * Ln + (size_t)(q0 + lh * 8) * Ln + l16;

  // persistent q A-operands (K = E = 64 split in two 32-chunks)
  const _Float16* qrow = qp + (size_t)(q0 + l16) * En;
  v16h qa0 = load_a16(qrow, aoff);
  v16h qa1 = load_a16(qrow + 32, aoff);

  v8f acc[4];
#pragma unroll
  for (int i = 0; i < 4; ++i) acc[i] = zero8();
  float m_[8], l_[8];
#pragma unroll
  for (int r = 0; r < 8; ++r) { m_[r] = -3.0e38f; l_[r] = 0.0f; }

  const float scale = 0.125f; // 1/sqrt(64)

  for (int kc = 0; kc < Ln; kc += 32) {
    v8f s[2];
#pragma unroll
    for (int t = 0; t < 2; ++t) {
      // S = q @ k^T : B operand is row-major k rows (K dim = e)
      const _Float16* krow = kp + (size_t)(kc + t * 16 + l16) * En;
      v16h kb0 = *(const v16h*)(krow + lh * 16);
      v16h kb1 = *(const v16h*)(krow + 32 + lh * 16);
      v8f tmp = WMMA_F16(qa0, kb0, zero8());
      s[t] = WMMA_F16(qa1, kb1, tmp);
      const float* bp = bp_base + kc + t * 16;
#pragma unroll
      for (int r = 0; r < 8; ++r) s[t][r] = s[t][r] * scale + bp[(size_t)r * Ln];
    }

    // online softmax over this 32-key chunk (row reductions across 16 lanes)
#pragma unroll
    for (int r = 0; r < 8; ++r) {
      float mx = fmaxf(s[0][r], s[1][r]);
      mx = fmaxf(mx, __shfl_xor(mx, 1));
      mx = fmaxf(mx, __shfl_xor(mx, 2));
      mx = fmaxf(mx, __shfl_xor(mx, 4));
      mx = fmaxf(mx, __shfl_xor(mx, 8));
      float mn = fmaxf(m_[r], mx);
      float rs = __expf(m_[r] - mn);
      float p0 = __expf(s[0][r] - mn);
      float p1 = __expf(s[1][r] - mn);
      s[0][r] = p0; s[1][r] = p1;
      float ps = p0 + p1;
      ps += __shfl_xor(ps, 1);
      ps += __shfl_xor(ps, 2);
      ps += __shfl_xor(ps, 4);
      ps += __shfl_xor(ps, 8);
      l_[r] = l_[r] * rs + ps;
      m_[r] = mn;
#pragma unroll
      for (int e = 0; e < 4; ++e) acc[e][r] *= rs;
    }

    // stash P (f16) in LDS row-major 16x32, then reload in A-layout
    {
      _Float16* pw = &pshm[wave][0];
#pragma unroll
      for (int r = 0; r < 8; ++r) {
        int row = lh * 8 + r;
        pw[row * 32 + l16]      = (_Float16)s[0][r];
        pw[row * 32 + 16 + l16] = (_Float16)s[1][r];
      }
    }
    asm volatile("s_wait_dscnt 0" ::: "memory"); // in-wave LDS store->load order

    v16h pa = load_a16(&pshm[wave][l16 * 32], aoff);
#pragma unroll
    for (int e = 0; e < 4; ++e) {
      // ctx += P @ V : B operand from vT rows (N = e, K = key) contiguous
      v16h vb = *(const v16h*)(vp + (size_t)(e * 16 + l16) * Ln + kc + lh * 16);
      acc[e] = WMMA_F16(pa, vb, acc[e]);
    }
  }

  // normalize and write ctx back as [B,L,D] f16
#pragma unroll
  for (int r = 0; r < 8; ++r) {
    float inv = 1.0f / l_[r];
    int l = q0 + lh * 8 + r;
#pragma unroll
    for (int e = 0; e < 4; ++e) {
      size_t idx = ((size_t)(b * Ln + l)) * Dn + hh * En + e * 16 + l16;
      ctx[idx] = (_Float16)(acc[e][r] * inv);
    }
  }
}

// ---------------------------------------------------------------------------
// LayerNorm over D=256 features. block = 256 threads = 1 row.
// v = p0 + (p1 ? p1 : 0); outputs optional f32 and f16.
// ---------------------------------------------------------------------------
__global__ __launch_bounds__(256) void k_layernorm(const float* __restrict__ p0,
                                                   const float* __restrict__ p1,
                                                   const float* __restrict__ g,
                                                   const float* __restrict__ bt,
                                                   float* __restrict__ o32,
                                                   _Float16* __restrict__ o16) {
  __shared__ float red[256];
  const int t = threadIdx.x;
  const size_t row = blockIdx.x;
  const size_t idx = row * Dn + t;

  float v = p0[idx] + (p1 ? p1[idx] : 0.0f);
  red[t] = v;
  __syncthreads();
#pragma unroll
  for (int s = 128; s > 0; s >>= 1) {
    if (t < s) red[t] += red[t + s];
    __syncthreads();
  }
  float mu = red[0] * (1.0f / 256.0f);
  __syncthreads();
  float d = v - mu;
  red[t] = d * d;
  __syncthreads();
#pragma unroll
  for (int s = 128; s > 0; s >>= 1) {
    if (t < s) red[t] += red[t + s];
    __syncthreads();
  }
  float var = red[0] * (1.0f / 256.0f);
  float y = d * rsqrtf(var + 1e-5f) * g[t] + bt[t];
  if (o32) o32[idx] = y;
  if (o16) o16[idx] = (_Float16)y;
}

// ---------------------------------------------------------------------------
// launch
// ---------------------------------------------------------------------------
extern "C" void kernel_launch(void* const* d_in, const int* in_sizes, int n_in,
                              void* d_out, int out_size, void* d_ws, size_t ws_size,
                              hipStream_t stream) {
  const float* x     = (const float*)d_in[0];
  const float* abias = (const float*)d_in[1];
  const float* Wq = (const float*)d_in[2];  const float* bq = (const float*)d_in[3];
  const float* Wk = (const float*)d_in[4];  const float* bk = (const float*)d_in[5];
  const float* Wv = (const float*)d_in[6];  const float* bv = (const float*)d_in[7];
  const float* Wo = (const float*)d_in[8];  const float* bo = (const float*)d_in[9];
  const float* g1 = (const float*)d_in[10]; const float* b1n = (const float*)d_in[11];
  const float* W1 = (const float*)d_in[12]; const float* b1 = (const float*)d_in[13];
  const float* W2 = (const float*)d_in[14]; const float* b2 = (const float*)d_in[15];
  const float* g2 = (const float*)d_in[16]; const float* b2n = (const float*)d_in[17];

  char* ws = (char*)d_ws;
  const size_t MB = 1ull << 20;
  _Float16* xh  = (_Float16*)(ws + 0);        // 8MB  (reused as h16 after QKV)
  _Float16* qb  = (_Float16*)(ws + 8 * MB);   // 8MB  \ reused together as
  _Float16* kb  = (_Float16*)(ws + 16 * MB);  // 8MB  / f1 (16MB) after attn
  _Float16* vtb = (_Float16*)(ws + 24 * MB);  // 8MB
  _Float16* ctx = (_Float16*)(ws + 32 * MB);  // 8MB
  float*    res1= (float*)(ws + 40 * MB);     // 16MB (reused as f2 after LN1)
  float*    h32 = (float*)(ws + 56 * MB);     // 16MB
  _Float16* WqT = (_Float16*)(ws + 72 * MB);  // ~1MB of weight transposes
  _Float16* WkT = WqT + 65536;
  _Float16* WvT = WkT + 65536;
  _Float16* WoT = WvT + 65536;
  _Float16* W1T = WoT + 65536;            // [F,D] = 512*256
  _Float16* W2T = W1T + Fn * Dn;          // [D,F] = 256*512
  _Float16* h16 = xh;
  _Float16* f1  = qb;
  float*    f2  = res1;

  // 1) precision/layout prep
  k_f32_to_f16<<<(Mn * Dn) / 256, 256, 0, stream>>>(x, xh, (size_t)Mn * Dn);
  k_transpose_f16<<<256, 256, 0, stream>>>(Wq, WqT, Dn, Dn);
  k_transpose_f16<<<256, 256, 0, stream>>>(Wk, WkT, Dn, Dn);
  k_transpose_f16<<<256, 256, 0, stream>>>(Wv, WvT, Dn, Dn);
  k_transpose_f16<<<256, 256, 0, stream>>>(Wo, WoT, Dn, Dn);
  k_transpose_f16<<<512, 256, 0, stream>>>(W1, W1T, Dn, Fn); // Wt[f, d]
  k_transpose_f16<<<512, 256, 0, stream>>>(W2, W2T, Fn, Dn); // Wt[d, f]

  dim3 gD(Dn / 64, Mn / 128); // (4,128)
  dim3 gF(Fn / 64, Mn / 128); // (8,128)

  // 2) QKV projections
  k_gemm_f16<<<gD, 256, 0, stream>>>(xh, WqT, bq, Dn, Dn, 0, nullptr, qb);
  k_gemm_f16<<<gD, 256, 0, stream>>>(xh, WkT, bk, Dn, Dn, 0, nullptr, kb);
  k_gemm_f16<<<gD, 256, 0, stream>>>(xh, WvT, bv, Dn, Dn, 1, nullptr, vtb);

  // 3) attention (flash-style, fused bias+softmax)
  k_attention<<<Bn * Hn * (Ln / 64), 128, 0, stream>>>(qb, kb, vtb, abias, ctx);

  // 4) output projection + residual
  k_gemm_f16<<<gD, 256, 0, stream>>>(ctx, WoT, bo, Dn, Dn, 2, x, res1);

  // 5) LN1 -> h (f32 + f16)
  k_layernorm<<<Mn, 256, 0, stream>>>(res1, nullptr, g1, b1n, h32, h16);

  // 6) FFN
  k_gemm_f16<<<gF, 256, 0, stream>>>(h16, W1T, b1, Dn, Fn, 3, nullptr, f1);
  k_gemm_f16<<<gD, 256, 0, stream>>>(f1, W2T, b2, Fn, Dn, 4, nullptr, f2);

  // 7) LN2 -> out
  k_layernorm<<<Mn, 256, 0, stream>>>(f2, h32, g2, b2n, (float*)d_out, nullptr);
}
